// Instance_DiceLoss_90821378441236
// MI455X (gfx1250) — matrix operand order repaired and checked by reading, not compiled, not run on hardware
//
#include <hip/hip_runtime.h>
#include <hip/hip_bf16.h>

typedef __attribute__((ext_vector_type(16))) _Float16 v16h;
typedef __attribute__((ext_vector_type(8)))  float    v8f;

#define NUM_GT   25
#define NUM_PRED 40
#define NPC      26            // NUM_GT+1  (gt labels 0..25)
#define NPP      41            // NUM_PRED+1 (pred labels 0..40)
#define NBINS    (NPP * NPC)   // 1066
#define HIST_COPIES 8          // one sub-histogram per wave32 in a 256-thread WG
#define HIST_STRIDE 1088       // padded bin stride

// ---------------------------------------------------------------------------
// Kernel 0: zero the global histogram (d_ws persists between replays, so we
// must re-init every launch for determinism).
// ---------------------------------------------------------------------------
__global__ void zero_hist(unsigned* __restrict__ g) {
  for (int i = threadIdx.x; i < NBINS; i += blockDim.x) g[i] = 0u;
}

// ---------------------------------------------------------------------------
// Kernel 1: joint histogram counts[p][c], p = round(pred), c = gt.
// HBM-bound: 134 MB streamed via b128 loads; LDS ds_add_u32 per-wave
// sub-histograms; one global_atomic_add_u32 merge per block per bin.
// ---------------------------------------------------------------------------
__global__ void __launch_bounds__(256)
joint_hist(const float* __restrict__ pred, const int* __restrict__ gt,
           unsigned* __restrict__ ghist, long long n) {
  __shared__ unsigned h[HIST_COPIES * HIST_STRIDE];
  const int tid = threadIdx.x;
  for (int i = tid; i < HIST_COPIES * HIST_STRIDE; i += 256) h[i] = 0u;
  __syncthreads();

  unsigned* mine = h + (tid >> 5) * HIST_STRIDE;  // wave32 -> tid/32 picks copy

  const long long n4     = n >> 2;
  const long long gid    = (long long)blockIdx.x * 256 + tid;
  const long long stride = (long long)gridDim.x * 256;

  const float4* p4 = reinterpret_cast<const float4*>(pred);
  const int4*   g4 = reinterpret_cast<const int4*>(gt);

  for (long long i = gid; i < n4; i += stride) {
    float4 pv = p4[i];
    int4   gv = g4[i];
    unsigned b0 = (unsigned)(__float2int_rn(pv.x) * NPC + gv.x);
    unsigned b1 = (unsigned)(__float2int_rn(pv.y) * NPC + gv.y);
    unsigned b2 = (unsigned)(__float2int_rn(pv.z) * NPC + gv.z);
    unsigned b3 = (unsigned)(__float2int_rn(pv.w) * NPC + gv.w);
    if (b0 < NBINS) atomicAdd(&mine[b0], 1u);
    if (b1 < NBINS) atomicAdd(&mine[b1], 1u);
    if (b2 < NBINS) atomicAdd(&mine[b2], 1u);
    if (b3 < NBINS) atomicAdd(&mine[b3], 1u);
  }
  // scalar tail (n not a multiple of 4)
  for (long long i = n4 * 4 + gid; i < n; i += stride) {
    unsigned b = (unsigned)(__float2int_rn(pred[i]) * NPC + gt[i]);
    if (b < NBINS) atomicAdd(&mine[b], 1u);
  }
  __syncthreads();

  for (int b = tid; b < NBINS; b += 256) {
    unsigned s = 0;
#pragma unroll
    for (int c = 0; c < HIST_COPIES; ++c) s += h[c * HIST_STRIDE + b];
    if (s) atomicAdd(&ghist[b], s);
  }
}

// ---------------------------------------------------------------------------
// Kernel 2: finalize with one wave32.
// union[c] = sum_p overlap[p][c] * pred_size[p] done as a WMMA matvec:
//   A (16x32 f16)  = overlap^T tile (0/1, exact in f16), K = pred label
//   B (32x16 f16)  = pred_size split into 8-bit digits in columns 0..2
//   D (f32)        = exact digit sums; recombine S0 + 256*S1 + 65536*S2.
// Operand builds read pre-padded f16 LDS tables -> branch-free, no exec
// toggling around the WMMA chain.  Element placement per ISA 7.12.2.
// ---------------------------------------------------------------------------
__global__ void __launch_bounds__(32)
finalize(const unsigned* __restrict__ ghist, float* __restrict__ out) {
  __shared__ unsigned  cnt[NBINS];
  __shared__ float     psf[64];        // pred_sizes (padded to 64)
  __shared__ _Float16  ovA[32][64];    // [j][p] overlap 0/1, padded with zeros
  __shared__ _Float16  bT[16][64];     // [n][p] digit table; rows 3..15 zero
  __shared__ float     gt_sz[32];
  __shared__ int       presentf[32];
  __shared__ float     dmat[2][16][16];
  __shared__ float     red_dice[32];
  __shared__ int       red_ng[32];
  __shared__ int       red_fp[32];

  const int lane   = threadIdx.x;
  const int laneHi = lane >> 4;   // 0: lanes 0-15, 1: lanes 16-31
  const int nCol   = lane & 15;   // N column / M row within tile

  for (int i = lane; i < NBINS; i += 32) cnt[i] = ghist[i];
  __syncthreads();

  // pred_sizes[p] = sum_c counts[p][c]  (exact in u32; <= 2^24) + digit split
  for (int p = lane; p < 64; p += 32) {
    unsigned s = 0;
    if (p < NPP)
      for (int c = 0; c < NPC; ++c) s += cnt[p * NPC + c];
    psf[p]   = (float)s;
    bT[0][p] = (_Float16)(float)(s & 255u);
    bT[1][p] = (_Float16)(float)((s >> 8) & 255u);
    bT[2][p] = (_Float16)(float)(s >> 16);      // <= 511, exact in f16
  }
  // zero rows 3..15 of the B table (N columns beyond the 3 digits)
  for (int i = lane; i < 13 * 64; i += 32)
    bT[3 + (i >> 6)][i & 63] = (_Float16)0.f;
  // padded overlap table: ovA[j][p] = counts[p][j+1] > 0 (else 0)
  for (int i = lane; i < 32 * 64; i += 32) {
    const int j = i >> 6, p = i & 63;
    float v = 0.f;
    if (j < NUM_GT && p < NPP)
      v = (cnt[p * NPC + j + 1] > 0u) ? 1.f : 0.f;
    ovA[j][p] = (_Float16)v;
  }
  // gt_sizes[j] = sum_p counts[p][j+1]
  {
    float g = 0.f;
    if (lane < NUM_GT)
      for (int p = 0; p < NPP; ++p) g += (float)cnt[p * NPC + (lane + 1)];
    gt_sz[lane]    = g;
    presentf[lane] = (lane < NUM_GT && g > 0.f) ? 1 : 0;
  }
  __syncthreads();

  // B operands (shared by both M tiles). 32x16 16-bit layout:
  // lanes 0-15 hold K in {0..7,16..23} (2 per VGPR), lanes 16-31 hold +8.
  v16h bk0, bk1;
#pragma unroll
  for (int h = 0; h < 16; ++h) {
    const int i = h >> 1, off = h & 1;
    const int K = (i < 4 ? 2 * i : 16 + 2 * (i - 4)) + off + (laneHi ? 8 : 0);
    bk0[h] = bT[nCol][K];        // kstep 0: p = K
    bk1[h] = bT[nCol][K + 32];   // kstep 1: p = K + 32
  }

#pragma unroll
  for (int t = 0; t < 2; ++t) {
    const int j = nCol + t * 16;  // gt component row held by this lane
    v8f acc = {};
#pragma unroll
    for (int ks = 0; ks < 2; ++ks) {
      // A 16x32 16-bit layout: lanes<16 K in {0..7,16..23}; lanes>=16 +8.
      v16h a;
#pragma unroll
      for (int h = 0; h < 16; ++h) {
        const int i = h >> 1, off = h & 1;
        const int K = (i < 4 ? 2 * i : 16 + 2 * (i - 4)) + off + (laneHi ? 8 : 0);
        a[h] = ovA[j][K + ks * 32];
      }
      acc = __builtin_amdgcn_wmma_f32_16x16x32_f16(
          /*neg_a=*/false, a, /*neg_b=*/false, (ks == 0) ? bk0 : bk1,
          /*c_mod=*/(short)0, acc, /*reuse_a=*/false, /*reuse_b=*/false);
    }
    // D layout: VGPR r holds M = r + 8*laneHi, N = lane&15
#pragma unroll
    for (int r = 0; r < 8; ++r) dmat[t][r + 8 * laneHi][nCol] = acc[r];
  }
  __syncthreads();

  // dice per gt component
  float dice = 0.f;
  const int pres = presentf[lane];
  if (lane < NUM_GT) {
    const int t = lane >> 4, m = lane & 15;
    const float u = dmat[t][m][0] + 256.f * dmat[t][m][1] + 65536.f * dmat[t][m][2];
    const float g = gt_sz[lane];
    if (pres) dice = 2.f * g / (u + g + 1.0f);  // SMOOTH = 1
  }
  red_dice[lane] = dice;
  red_ng[lane]   = pres;

  // false positives: pred labels with voxels but no overlap with present comps
  int fp = 0;
  for (int p = lane; p < NPP; p += 32) {
    if (psf[p] > 0.f) {
      int tp = 0;
      for (int jj = 0; jj < NUM_GT; ++jj)
        if (presentf[jj] && cnt[p * NPC + (jj + 1)] > 0u) { tp = 1; break; }
      if (!tp) fp++;
    }
  }
  red_fp[lane] = fp;
  __syncthreads();

  if (lane == 0) {
    float ld = 0.f; int ng = 0, fps = 0;
    for (int i = 0; i < 32; ++i) { ld += red_dice[i]; ng += red_ng[i]; fps += red_fp[i]; }
    out[0] = ld / (float)(ng + fps);
  }
}

// ---------------------------------------------------------------------------
extern "C" void kernel_launch(void* const* d_in, const int* in_sizes, int n_in,
                              void* d_out, int out_size, void* d_ws, size_t ws_size,
                              hipStream_t stream) {
  const float* pred  = (const float*)d_in[0];
  const int*   gt    = (const int*)d_in[1];
  unsigned*    ghist = (unsigned*)d_ws;   // 1066 * 4 B scratch
  float*       outp  = (float*)d_out;
  const long long n  = (long long)in_sizes[0];

  zero_hist <<<1, 256, 0, stream>>>(ghist);
  joint_hist<<<512, 256, 0, stream>>>(pred, gt, ghist, n);
  finalize  <<<1, 32, 0, stream>>>(ghist, outp);
}
